// RGCNModule_60962765799960
// MI455X (gfx1250) — matrix operand order repaired and checked by reading, not compiled, not run on hardware
//
#include <hip/hip_runtime.h>
#include <hip/hip_bf16.h>

typedef __attribute__((ext_vector_type(16))) _Float16 v16h;
typedef __attribute__((ext_vector_type(8)))  float    v8f;

#define N_NODES 50000
#define N_EDGES 800000
#define IN_DIM  128
#define HID     64
#define NCOLS   192   // [root | W_r0 | W_r1] fused output columns
#define LN_EPS  1e-5f

// ---------------------------------------------------------------------------
// Fused WMMA GEMM: C[M,192] = A[M,K] @ [root | W[0] | W[1]]   (f32 in, f16
// multiply via V_WMMA_F32_16X16X32_F16, f32 accumulate).
// One wave per 16-row block; A fragments are loaded/converted ONCE and reused
// across all 12 N-tiles (registers: K/32 x v16h). K is a template parameter so
// the chunk loop fully unrolls into back-to-back WMMAs. grid = (M/16).
// A-fragment k order chosen here (B matches): lane m(0-15): K = k0+[0..15]
//                                     lane m+16  : K = k0+16+[0..15]
// B-fragment k order chosen here (A matches): lanes 0-15 hold K=k0+[0..15] of column n,
//                                     lanes 16-31 hold K=k0+16+[0..15].
// D layout (f32 16x16): VGPR i -> (M=i, N=lane) / (M=8+i, N=lane-16).
// ---------------------------------------------------------------------------
template<int K>
__global__ __launch_bounds__(32) void rgcn_gemm_wmma(
    const float* __restrict__ A,
    const float* __restrict__ root,   // [K,64]
    const float* __restrict__ W,      // [2,K,64]
    float* __restrict__ C)            // [M,192]
{
    constexpr int NCH = K / 32;
    const int lane  = threadIdx.x;
    const int n15   = lane & 15;
    const int hi    = lane >> 4;
    const int mBase = blockIdx.x * 16;

    const float* Arow = A + (size_t)(mBase + n15) * K;

    // Load + convert all A fragments for this row block once.
    v16h afrag[NCH];
    #pragma unroll
    for (int kc = 0; kc < NCH; ++kc) {
        const int k0 = kc << 5;
        const float4* ap0 = (const float4*)(Arow + k0 + hi * 16);
        const float4* ap1 = (const float4*)(Arow + k0 + hi * 16 + 8);
        float4 f0 = ap0[0], f1 = ap0[1], f2 = ap1[0], f3 = ap1[1];
        v16h a;
        a[0]=(_Float16)f0.x;  a[1]=(_Float16)f0.y;  a[2]=(_Float16)f0.z;  a[3]=(_Float16)f0.w;
        a[4]=(_Float16)f1.x;  a[5]=(_Float16)f1.y;  a[6]=(_Float16)f1.z;  a[7]=(_Float16)f1.w;
        a[8]=(_Float16)f2.x;  a[9]=(_Float16)f2.y;  a[10]=(_Float16)f2.z; a[11]=(_Float16)f2.w;
        a[12]=(_Float16)f3.x; a[13]=(_Float16)f3.y; a[14]=(_Float16)f3.z; a[15]=(_Float16)f3.w;
        afrag[kc] = a;
    }

    // Sweep the 12 output tiles (root cols 0..63, W[0] cols 64..127, W[1] 128..191)
    for (int nTile = 0; nTile < NCOLS / 16; ++nTile) {
        const int seg = nTile >> 2;            // 0=root, 1=W[0], 2=W[1]
        const int nc0 = (nTile & 3) * 16;
        const float* Bsel = (seg == 0) ? root : (W + (size_t)(seg - 1) * K * 64);

        v8f acc = {};
        #pragma unroll
        for (int kc = 0; kc < NCH; ++kc) {
            const int k0 = kc << 5;
            const float* bp = Bsel + (size_t)(k0 + hi * 16) * 64 + nc0 + n15;
            v16h b;
            #pragma unroll
            for (int j = 0; j < 16; ++j) b[j] = (_Float16)bp[(size_t)j * 64];
            acc = __builtin_amdgcn_wmma_f32_16x16x32_f16(
                      /*neg_a=*/false, afrag[kc], /*neg_b=*/false, b,
                      /*c_mod=*/(short)0, acc, /*reuse_a=*/false, /*reuse_b=*/false);
        }

        float* crow = C + (size_t)mBase * NCOLS + nTile * 16 + n15;
        #pragma unroll
        for (int i = 0; i < 8; ++i)
            crow[(size_t)(i + hi * 8) * NCOLS] = acc[i];
    }
}

// ---------------------------------------------------------------------------
__global__ __launch_bounds__(256) void zero_f32(float* __restrict__ p, long long n)
{
    long long i = (long long)blockIdx.x * blockDim.x + threadIdx.x;
    if (i < n) p[i] = 0.0f;
}

// Per-relation in-degree counts: cnt[r*N + dst] += 1  (one thread per edge)
__global__ __launch_bounds__(256) void count_edges(
    const long long* __restrict__ dst,
    const long long* __restrict__ etype,
    float* __restrict__ cnt)
{
    int e = blockIdx.x * blockDim.x + threadIdx.x;
    if (e < N_EDGES) {
        int r = (int)etype[e];
        atomicAdd(&cnt[(size_t)r * N_NODES + (size_t)dst[e]], 1.0f);
    }
}

// Scatter-add: agg[r][dst] += proj_r[src].  proj_r lives in C columns 64+64r.
// Thread mapping: (edge, feature) with feature minor -> coalesced gather and
// a contiguous 64-float atomic destination row per edge (L2-resident).
__global__ __launch_bounds__(256) void scatter_edges(
    const float* __restrict__ C,       // [N,192]
    const long long* __restrict__ src,
    const long long* __restrict__ dst,
    const long long* __restrict__ etype,
    float* __restrict__ agg)           // [2,N,64]
{
    long long gid = (long long)blockIdx.x * blockDim.x + threadIdx.x;
    if (gid >= (long long)N_EDGES * 64) return;
    int e = (int)(gid >> 6);
    int f = (int)(gid & 63);
    int r = (int)etype[e];
    float val = C[(size_t)src[e] * NCOLS + 64 + r * 64 + f];
    atomicAdd(&agg[((size_t)r * N_NODES + (size_t)dst[e]) * 64 + f], val);
}

// Combine self + mean-aggregated messages, LayerNorm, optional exact GELU.
// One wave per node (wave32: 2 features per lane), shfl reductions.
__global__ __launch_bounds__(256) void combine_ln(
    const float* __restrict__ C,       // [N,192] (cols 0..63 = x@root)
    const float* __restrict__ agg,     // [2,N,64]
    const float* __restrict__ cnt,     // [2,N]
    const float* __restrict__ bias,
    const float* __restrict__ gamma,
    const float* __restrict__ beta,
    float* __restrict__ out,           // [N,64]
    int doGelu)
{
    int wave = threadIdx.x >> 5;
    int lane = threadIdx.x & 31;
    int node = blockIdx.x * 8 + wave;
    if (node >= N_NODES) return;

    float inv0 = 1.0f / fmaxf(cnt[node], 1.0f);
    float inv1 = 1.0f / fmaxf(cnt[(size_t)N_NODES + node], 1.0f);

    float v[2];
    #pragma unroll
    for (int t = 0; t < 2; ++t) {
        int f = lane + t * 32;
        float s = C[(size_t)node * NCOLS + f] + bias[f];
        s += agg[(size_t)node * 64 + f] * inv0;
        s += agg[((size_t)N_NODES + node) * 64 + f] * inv1;
        v[t] = s;
    }

    float sum = v[0] + v[1];
    #pragma unroll
    for (int off = 16; off; off >>= 1) sum += __shfl_xor(sum, off, 32);
    float mu = sum * (1.0f / 64.0f);

    float d0 = v[0] - mu, d1 = v[1] - mu;
    float vs = d0 * d0 + d1 * d1;
    #pragma unroll
    for (int off = 16; off; off >>= 1) vs += __shfl_xor(vs, off, 32);
    float rstd = rsqrtf(vs * (1.0f / 64.0f) + LN_EPS);

    #pragma unroll
    for (int t = 0; t < 2; ++t) {
        int f = lane + t * 32;
        float y = (v[t] - mu) * rstd * gamma[f] + beta[f];
        if (doGelu) y = 0.5f * y * (1.0f + erff(y * 0.70710678118654752f));
        out[(size_t)node * 64 + f] = y;
    }
}

// ---------------------------------------------------------------------------
extern "C" void kernel_launch(void* const* d_in, const int* in_sizes, int n_in,
                              void* d_out, int out_size, void* d_ws, size_t ws_size,
                              hipStream_t stream) {
    const float*     x     = (const float*)d_in[0];
    const long long* eidx  = (const long long*)d_in[1];   // [2,E] int64
    const long long* etype = (const long long*)d_in[2];   // [E]   int64
    const float*     W1    = (const float*)d_in[3];
    const float*     root1 = (const float*)d_in[4];
    const float*     b1    = (const float*)d_in[5];
    const float*     g1    = (const float*)d_in[6];
    const float*     beta1 = (const float*)d_in[7];
    const float*     W2    = (const float*)d_in[8];
    const float*     root2 = (const float*)d_in[9];
    const float*     b2    = (const float*)d_in[10];
    const float*     g2    = (const float*)d_in[11];
    const float*     beta2 = (const float*)d_in[12];
    float*           out   = (float*)d_out;

    const long long* src = eidx;
    const long long* dst = eidx + N_EDGES;

    // Workspace layout (floats):
    //  Cbuf: N*192 = 9,600,000   (reused by both layers)
    //  agg : 2*N*64 = 6,400,000
    //  cnt : 2*N    =   100,000
    //  h1  : N*64   = 3,200,000
    float* Cbuf = (float*)d_ws;
    float* agg  = Cbuf + (size_t)N_NODES * NCOLS;
    float* cnt  = agg  + (size_t)2 * N_NODES * 64;
    float* h1   = cnt  + (size_t)2 * N_NODES;

    const int gemmBlocks     = N_NODES / 16;             // 3125 row blocks
    const long long nScatter = (long long)N_EDGES * 64;
    const int scatterBlocks  = (int)((nScatter + 255) / 256);
    const int combineBlocks  = (N_NODES + 7) / 8;

    // agg and cnt are contiguous: zero both in one pass
    long long nZeroAll = (long long)2 * N_NODES * 64 + 2 * N_NODES;
    zero_f32<<<(int)((nZeroAll + 255) / 256), 256, 0, stream>>>(agg, nZeroAll);

    count_edges<<<(N_EDGES + 255) / 256, 256, 0, stream>>>(dst, etype, cnt);

    // ---- Layer 1 ----
    rgcn_gemm_wmma<IN_DIM><<<gemmBlocks, 32, 0, stream>>>(x, root1, W1, Cbuf);
    scatter_edges<<<scatterBlocks, 256, 0, stream>>>(Cbuf, src, dst, etype, agg);
    combine_ln<<<combineBlocks, 256, 0, stream>>>(Cbuf, agg, cnt, b1, g1, beta1, h1, 1);

    // ---- Layer 2 ----
    long long nZeroAgg = (long long)2 * N_NODES * 64;
    zero_f32<<<(int)((nZeroAgg + 255) / 256), 256, 0, stream>>>(agg, nZeroAgg);
    rgcn_gemm_wmma<HID><<<gemmBlocks, 32, 0, stream>>>(h1, root2, W2, Cbuf);
    scatter_edges<<<scatterBlocks, 256, 0, stream>>>(Cbuf, src, dst, etype, agg);
    combine_ln<<<combineBlocks, 256, 0, stream>>>(Cbuf, agg, cnt, b2, g2, beta2, out, 0);
}
